// RNN_39994735460584
// MI455X (gfx1250) — matrix operand order repaired and checked by reading, not compiled, not run on hardware
//
#include <hip/hip_runtime.h>
#include <hip/hip_bf16.h>

typedef __attribute__((ext_vector_type(16))) _Float16 v16h;
typedef __attribute__((ext_vector_type(8)))  _Float16 v8h;
typedef __attribute__((ext_vector_type(8)))  float    v8f;

union AF { v16h v; v8h h[2]; };

// fast tanh: 1 - 2/(e^{2x}+1); saturates correctly at +/-inf. One v_exp_f32 + v_rcp.
__device__ __forceinline__ float fast_tanh(float x) {
  float e = __expf(2.0f * x);
  return 1.0f - 2.0f / (e + 1.0f);
}

// ---------------------------------------------------------------------------
// Fragment loaders (CDNA5 WMMA 16x16x32 f16 register layouts, ISA 7.12.2)
// A (16x32, M x K), row-major source with row stride `stride` (halfs):
//   lanes 0-15 : elems 0..7 = K k0+0..7,  elems 8..15 = K k0+16..23, row M=lane
//   lanes 16-31: elems 0..7 = K k0+8..15, elems 8..15 = K k0+24..31, row M=lane-16
// ---------------------------------------------------------------------------
__device__ __forceinline__ v16h load_afrag(const _Float16* base, int stride,
                                           int row, int k0, int lane) {
  int kb = k0 + ((lane >> 4) << 3);
  const _Float16* p = base + (size_t)row * stride + kb;
  AF f;
  f.h[0] = *(const v8h*)(p);
  f.h[1] = *(const v8h*)(p + 16);
  return f.v;
}

// B (32x16, K x N), source is W^T row-major [N][stride] (halfs):
//   lanes 0-15 : elems 0..7 = K k0+0..7,  elems 8..15 = K k0+16..23, col N = lane
//   lanes 16-31: elems 0..7 = K k0+8..15, elems 8..15 = K k0+24..31, col N = lane-16
__device__ __forceinline__ v16h load_bfrag(const _Float16* wT, int stride,
                                           int col, int k0, int lane) {
  int kb = k0 + ((lane >> 4) << 3);
  const _Float16* p = wT + (size_t)col * stride + kb;
  AF f;
  f.h[0] = *(const v8h*)(p);
  f.h[1] = *(const v8h*)(p + 16);
  return f.v;
}

// ---------------------------------------------------------------------------
// Pack kernels: f32 -> f16 transposed weights resident in L2
// ---------------------------------------------------------------------------
__global__ void k_pack_weT(const float* __restrict__ We, _Float16* __restrict__ WeT) {
  int i = blockIdx.x * 256 + threadIdx.x;           // 512 x 320
  if (i >= 512 * 320) return;
  int n = i / 320, k = i % 320;
  WeT[i] = (_Float16)((k < 300) ? We[(size_t)k * 512 + n] : 0.0f);
}

__global__ void k_pack_wl(const float* __restrict__ wx, const float* __restrict__ wh,
                          _Float16* __restrict__ WT) {
  int i = blockIdx.x * 256 + threadIdx.x;           // 512 x 1024
  if (i >= 512 * 1024) return;
  int n = i / 1024, k = i % 1024;
  float v = (k < 512) ? wx[(size_t)k * 512 + n] : wh[(size_t)(k - 512) * 512 + n];
  WT[i] = (_Float16)v;
}

__global__ void k_bsum(const float* __restrict__ bx, const float* __restrict__ bh,
                       float* __restrict__ bs) {
  int i = blockIdx.x * 256 + threadIdx.x;
  if (i < 512) bs[i] = bx[i] + bh[i];
}

// ---------------------------------------------------------------------------
// Embedding: emb[t][b][h] = f16( x[b][t][:300] @ We + be )
// block: 64 rows (b) x 512 cols, blockIdx.y = t
// dynamic LDS: 64 KB, used first as sX[64][320] f16, then as sOut[64][512] f16
// ---------------------------------------------------------------------------
__global__ __launch_bounds__(256)
void k_embed(const float* __restrict__ x, const _Float16* __restrict__ WeT,
             const float* __restrict__ be, _Float16* __restrict__ emb) {
  const int t = blockIdx.y;
  const int base = blockIdx.x * 64;
  extern __shared__ char smem_e[];
  _Float16* sX = (_Float16*)smem_e;                 // 64*320 halfs (40 KB)

  for (int idx = threadIdx.x; idx < 64 * 320; idx += 256) {
    int r = idx / 320, k = idx % 320;
    float v = (k < 300) ? x[((size_t)(base + r) * 20 + t) * 300 + k] : 0.0f;
    sX[idx] = (_Float16)v;
  }
  __syncthreads();

  const int lane = threadIdx.x & 31;
  const int w    = threadIdx.x >> 5;
  const int colBase = w * 64;

  v8f acc[4][4] = {};
  for (int ks = 0; ks < 10; ++ks) {
    int k0 = ks * 32;
    v16h bfr[4], afr[4];
#pragma unroll
    for (int nt = 0; nt < 4; ++nt)
      bfr[nt] = load_bfrag(WeT, 320, colBase + nt * 16 + (lane & 15), k0, lane);
#pragma unroll
    for (int mt = 0; mt < 4; ++mt)
      afr[mt] = load_afrag(sX, 320, mt * 16 + (lane & 15), k0, lane);
#pragma unroll
    for (int mt = 0; mt < 4; ++mt)
#pragma unroll
      for (int nt = 0; nt < 4; ++nt)
        acc[mt][nt] = __builtin_amdgcn_wmma_f32_16x16x32_f16(
            false, afr[mt], false, bfr[nt], (short)0, acc[mt][nt], false, false);
  }
  __syncthreads();                                   // done with sX; reuse as sOut

  _Float16* sOut = (_Float16*)smem_e;                // 64*512 halfs (64 KB)
  const int halfSel = lane >> 4;
#pragma unroll
  for (int nt = 0; nt < 4; ++nt) {
    int n = colBase + nt * 16 + (lane & 15);
    float bn = be[n];
#pragma unroll
    for (int mt = 0; mt < 4; ++mt)
#pragma unroll
      for (int i = 0; i < 8; ++i) {
        int r = mt * 16 + i + 8 * halfSel;
        sOut[r * 512 + n] = (_Float16)(acc[mt][nt][i] + bn);
      }
  }
  __syncthreads();

  // coalesced b128 stream-out
  const v8h* s = (const v8h*)sOut;
  v8h* d = (v8h*)(emb + ((size_t)t * 8192 + base) * 512);
  for (int idx = threadIdx.x; idx < 64 * 512 / 8; idx += 256) d[idx] = s[idx];
}

// ---------------------------------------------------------------------------
// Persistent scan: each block owns 64 batch rows, full hidden state in LDS.
// ---------------------------------------------------------------------------
struct ScanParams {
  const _Float16* emb;
  const _Float16* WT0; const _Float16* WT1; const _Float16* WT2;
  const float* bsum0; const float* bsum1; const float* bsum2;
  const float* g0; const float* g1; const float* g2;
  const float* beta0; const float* beta1; const float* beta2;
  const float* wfc; const float* bfc;
  float* out;
};

__global__ __launch_bounds__(256)
void k_scan(ScanParams P) {
  const int base = blockIdx.x * 64;
  const int tid  = threadIdx.x;
  const int lane = tid & 31;
  const int w    = tid >> 5;
  const int colBase = w * 64;
  const int halfSel = lane >> 4;

  extern __shared__ char smem[];
  _Float16* sV  = (_Float16*)smem;          // 64x512
  _Float16* sH0 = sV  + 64 * 512;           // 64x512 x3 (contiguous)
  _Float16* sH1 = sH0 + 64 * 512;
  _Float16* sH2 = sH1 + 64 * 512;
  float* sSum  = (float*)(sH2 + 64 * 512);  // [64][8]
  float* sSq   = sSum + 64 * 8;             // [64][8]
  float* sMean = sSq  + 64 * 8;             // [64]
  float* sRstd = sMean + 64;                // [64]

  // zero h0,h1,h2
  for (int idx = tid; idx < 3 * 64 * 512; idx += 256) sH0[idx] = (_Float16)0.0f;

  const _Float16* WTl[3]  = {P.WT0, P.WT1, P.WT2};
  const float* bsuml[3]   = {P.bsum0, P.bsum1, P.bsum2};
  const float* gl[3]      = {P.g0, P.g1, P.g2};
  const float* betal[3]   = {P.beta0, P.beta1, P.beta2};
  _Float16* hs[3]         = {sH0, sH1, sH2};

  // per-lane persistent LN params for this wave's 4 column tiles
  float gp[3][4], bp[3][4], bb[3][4];
#pragma unroll
  for (int l = 0; l < 3; ++l)
#pragma unroll
    for (int nt = 0; nt < 4; ++nt) {
      int n = colBase + nt * 16 + (lane & 15);
      gp[l][nt] = gl[l][n];
      bp[l][nt] = betal[l][n];
      bb[l][nt] = bsuml[l][n];
    }

  for (int t = 0; t < 20; ++t) {
    // stage v_t = emb[t][base..base+63][:] into LDS (coalesced b128)
    const v8h* src = (const v8h*)(P.emb + ((size_t)t * 8192 + base) * 512);
    v8h* dst = (v8h*)sV;
    for (int idx = tid; idx < 64 * 512 / 8; idx += 256) dst[idx] = src[idx];
    __syncthreads();

    // prefetch next timestep's 64 KB slab into cache while we compute
    if (t < 19) {
      const char* nxt = (const char*)(P.emb + ((size_t)(t + 1) * 8192 + base) * 512);
      for (int off = tid * 128; off < 64 * 512 * 2; off += 256 * 128)
        __builtin_prefetch(nxt + off, 0, 3);
    }

    const _Float16* in = sV;
    for (int l = 0; l < 3; ++l) {
      _Float16* hcur = hs[l];
      const _Float16* WT = WTl[l];

      v8f acc[4][4] = {};
      for (int ks = 0; ks < 32; ++ks) {
        int k0 = ks * 32;
        const _Float16* Ab = (k0 < 512) ? in : hcur;
        int kk = (k0 < 512) ? k0 : (k0 - 512);
        v16h bfr[4], afr[4];
#pragma unroll
        for (int nt = 0; nt < 4; ++nt)
          bfr[nt] = load_bfrag(WT, 1024, colBase + nt * 16 + (lane & 15), k0, lane);
#pragma unroll
        for (int mt = 0; mt < 4; ++mt)
          afr[mt] = load_afrag(Ab, 512, mt * 16 + (lane & 15), kk, lane);
#pragma unroll
        for (int mt = 0; mt < 4; ++mt)
#pragma unroll
          for (int nt = 0; nt < 4; ++nt)
            acc[mt][nt] = __builtin_amdgcn_wmma_f32_16x16x32_f16(
                false, afr[mt], false, bfr[nt], (short)0, acc[mt][nt], false, false);
      }

      // add biases (part of z, so must precede the mean)
#pragma unroll
      for (int mt = 0; mt < 4; ++mt)
#pragma unroll
        for (int nt = 0; nt < 4; ++nt)
#pragma unroll
          for (int i = 0; i < 8; ++i) acc[mt][nt][i] += bb[l][nt];

      // per-row partial sums over this wave's 64 columns
#pragma unroll
      for (int mt = 0; mt < 4; ++mt)
#pragma unroll
        for (int i = 0; i < 8; ++i) {
          float s = 0.f, s2 = 0.f;
#pragma unroll
          for (int nt = 0; nt < 4; ++nt) {
            float z = acc[mt][nt][i];
            s += z; s2 += z * z;
          }
          // reduce across the 16 lanes holding the same row (stay within half)
#pragma unroll
          for (int m = 1; m <= 8; m <<= 1) {
            s  += __shfl_xor(s,  m, 32);
            s2 += __shfl_xor(s2, m, 32);
          }
          if ((lane & 15) == 0) {
            int row = mt * 16 + i + 8 * halfSel;
            sSum[row * 8 + w] = s;
            sSq[row * 8 + w]  = s2;
          }
        }
      __syncthreads();

      if (tid < 64) {
        float s = 0.f, s2 = 0.f;
#pragma unroll
        for (int ww = 0; ww < 8; ++ww) { s += sSum[tid * 8 + ww]; s2 += sSq[tid * 8 + ww]; }
        float mean = s * (1.0f / 512.0f);
        float var  = s2 * (1.0f / 512.0f) - mean * mean;
        sMean[tid] = mean;
        sRstd[tid] = rsqrtf(var + 1e-5f);
      }
      __syncthreads();

      // apply LN + fast tanh, write new h (f16) into LDS
#pragma unroll
      for (int mt = 0; mt < 4; ++mt)
#pragma unroll
        for (int i = 0; i < 8; ++i) {
          int row = mt * 16 + i + 8 * halfSel;
          float mean = sMean[row], rs = sRstd[row];
#pragma unroll
          for (int nt = 0; nt < 4; ++nt) {
            int n = colBase + nt * 16 + (lane & 15);
            float hv = fast_tanh((acc[mt][nt][i] - mean) * rs * gp[l][nt] + bp[l][nt]);
            hcur[row * 512 + n] = (_Float16)hv;
          }
        }
      __syncthreads();
      in = hcur;
    }
  }

  // stage wfc (512x10 f32 = 20 KB) into the now-dead sV region
  float* sWfc = (float*)sV;
  for (int i = tid; i < 512 * 10; i += 256) sWfc[i] = P.wfc[i];
  __syncthreads();

  // final FC: logits = h2 @ wfc + bfc   (64 rows x 10 classes, K=512)
  for (int p = tid; p < 64 * 10; p += 256) {
    int r = p / 10, c = p % 10;
    float s = P.bfc[c];
    for (int k = 0; k < 512; ++k)
      s += (float)sH2[r * 512 + k] * sWfc[k * 10 + c];
    P.out[(size_t)(base + r) * 10 + c] = s;
  }
}

// ---------------------------------------------------------------------------
// Host launcher
// ---------------------------------------------------------------------------
extern "C" void kernel_launch(void* const* d_in, const int* in_sizes, int n_in,
                              void* d_out, int out_size, void* d_ws, size_t ws_size,
                              hipStream_t stream) {
  const float* x   = (const float*)d_in[0];
  const float* We  = (const float*)d_in[1];
  const float* be  = (const float*)d_in[2];
  const float* wfc = (const float*)d_in[3];
  const float* bfc = (const float*)d_in[4];
  const float* wx[3] = {(const float*)d_in[5],  (const float*)d_in[11], (const float*)d_in[17]};
  const float* bx[3] = {(const float*)d_in[6],  (const float*)d_in[12], (const float*)d_in[18]};
  const float* wh[3] = {(const float*)d_in[7],  (const float*)d_in[13], (const float*)d_in[19]};
  const float* bh[3] = {(const float*)d_in[8],  (const float*)d_in[14], (const float*)d_in[20]};
  const float* g[3]  = {(const float*)d_in[9],  (const float*)d_in[15], (const float*)d_in[21]};
  const float* bt[3] = {(const float*)d_in[10], (const float*)d_in[16], (const float*)d_in[22]};

  char* ws = (char*)d_ws;
  size_t off = 0;
  _Float16* WeT = (_Float16*)(ws + off); off += (size_t)512 * 320 * 2;      // 327,680
  _Float16* WT[3];
  for (int l = 0; l < 3; ++l) { WT[l] = (_Float16*)(ws + off); off += (size_t)512 * 1024 * 2; }
  float* bsum[3];
  for (int l = 0; l < 3; ++l) { bsum[l] = (float*)(ws + off); off += (size_t)512 * 4; }
  _Float16* emb = (_Float16*)(ws + off); off += (size_t)20 * 8192 * 512 * 2; // ~168 MB

  // pack weights (f16, transposed, L2-resident)
  k_pack_weT<<<(512 * 320 + 255) / 256, 256, 0, stream>>>(We, WeT);
  for (int l = 0; l < 3; ++l)
    k_pack_wl<<<(512 * 1024 + 255) / 256, 256, 0, stream>>>(wx[l], wh[l], WT[l]);
  for (int l = 0; l < 3; ++l)
    k_bsum<<<2, 256, 0, stream>>>(bx[l], bh[l], bsum[l]);

  // embedding GEMM (64 KB dynamic LDS)
  const int embLds = 64 * 512 * 2;
  (void)hipFuncSetAttribute((const void*)k_embed,
                            hipFuncAttributeMaxDynamicSharedMemorySize, embLds);
  dim3 ge(128, 20);
  k_embed<<<ge, 256, embLds, stream>>>(x, WeT, be, emb);

  // persistent scan (266,752 B dynamic LDS; WGP supports 320 KB)
  const int scanLds = 4 * 64 * 512 * 2 + (64 * 8 * 2 + 128) * 4;
  (void)hipFuncSetAttribute((const void*)k_scan,
                            hipFuncAttributeMaxDynamicSharedMemorySize, scanLds);
  ScanParams P;
  P.emb = emb;
  P.WT0 = WT[0]; P.WT1 = WT[1]; P.WT2 = WT[2];
  P.bsum0 = bsum[0]; P.bsum1 = bsum[1]; P.bsum2 = bsum[2];
  P.g0 = g[0]; P.g1 = g[1]; P.g2 = g[2];
  P.beta0 = bt[0]; P.beta1 = bt[1]; P.beta2 = bt[2];
  P.wfc = wfc; P.bfc = bfc;
  P.out = (float*)d_out;
  k_scan<<<128, 256, scanLds, stream>>>(P);
}
